// XoronAttention_87445534147206
// MI455X (gfx1250) — matrix operand order repaired and checked by reading, not compiled, not run on hardware
//
#include <hip/hip_runtime.h>

typedef __attribute__((ext_vector_type(16))) _Float16 v16h;
typedef __attribute__((ext_vector_type(8)))  _Float16 v8h;
typedef __attribute__((ext_vector_type(8)))  float    v8f;
typedef __attribute__((ext_vector_type(4)))  int      v4i;

#define S_LEN   4096
#define HID     2048
#define HEADS   16
#define KVHEADS 4
#define HD      128

#if __has_builtin(__builtin_amdgcn_global_load_async_to_lds_b128) && \
    __has_builtin(__builtin_amdgcn_s_wait_asynccnt)
#define HAVE_ASYNC_LDS 1
typedef __attribute__((address_space(1))) v4i g_v4i;   // global (device) AS
typedef __attribute__((address_space(3))) v4i l_v4i;   // LDS AS
#else
#define HAVE_ASYNC_LDS 0
#endif

static __device__ __forceinline__ v16h cat8(v8h lo, v8h hi) {
  return __builtin_shufflevector(lo, hi, 0,1,2,3,4,5,6,7,8,9,10,11,12,13,14,15);
}
static __device__ __forceinline__ v8f wmma_f16(v16h a, v16h b, v8f c) {
  // D = A(16x32 f16) * B(32x16 f16) + C(16x16 f32)
  return __builtin_amdgcn_wmma_f32_16x16x32_f16(false, a, false, b, (short)0, c, false, false);
}

// 32-byte global->LDS stage (async if available)
static __device__ __forceinline__ void stage32(const _Float16* g, _Float16* l) {
#if HAVE_ASYNC_LDS
  __builtin_amdgcn_global_load_async_to_lds_b128((g_v4i*)g, (l_v4i*)l, 0,  0);
  __builtin_amdgcn_global_load_async_to_lds_b128((g_v4i*)g, (l_v4i*)l, 16, 0);
#else
  *(v16h*)l = *(const v16h*)g;
#endif
}
static __device__ __forceinline__ void stage_wait() {
#if HAVE_ASYNC_LDS
  __builtin_amdgcn_s_wait_asynccnt(0);
#endif
}

// ---------------------------------------------------------------- converts
__global__ void k_f32_to_f16(const float* __restrict__ in, _Float16* __restrict__ out, int n) {
  int i = blockIdx.x * blockDim.x + threadIdx.x;
  if (i < n) out[i] = (_Float16)in[i];
}

// W[K][N] (f32) -> Wt[N][K] (f16)
__global__ void k_transpose_f16(const float* __restrict__ W, _Float16* __restrict__ Wt,
                                int K, int N) {
  int i = blockIdx.x * blockDim.x + threadIdx.x;
  if (i < K * N) {
    int k = i / N, n = i - k * N;
    Wt[(size_t)n * K + k] = (_Float16)W[i];
  }
}

// ---------------------------------------------------------------- GEMM
// C[M,N] = A[M,K] (f16 row major) * Bt[N,K] (f16, B pre-transposed)
// one wave computes a 32x64 tile (2 M-subtiles x 4 N-subtiles, 8 WMMA accs);
// 4 waves per block stack along M (block tile 128xM).
__global__ __launch_bounds__(128) void k_gemm(const _Float16* __restrict__ A,
                                              const _Float16* __restrict__ Bt,
                                              float* __restrict__ Cf,
                                              _Float16* __restrict__ Ch,
                                              int M, int N, int K) {
  const int lane = threadIdx.x & 31;
  const int wid  = threadIdx.x >> 5;
  const int m0 = blockIdx.x * 128 + wid * 32;
  const int n0 = blockIdx.y * 64;
  const int nn = lane & 15;
  const int hi = lane >> 4;

  v8f z = {};
  v8f acc[2][4];
  #pragma unroll
  for (int mi = 0; mi < 2; ++mi)
    #pragma unroll
    for (int t = 0; t < 4; ++t) acc[mi][t] = z;

  const _Float16* arow0 = A + (size_t)(m0 + nn) * K;
  const _Float16* arow1 = A + (size_t)(m0 + 16 + nn) * K;
  for (int kb = 0; kb < K; kb += 32) {
    v16h a0 = cat8(*(const v8h*)(arow0 + kb + 16 * hi),
                   *(const v8h*)(arow0 + kb + 16 * hi + 8));
    v16h a1 = cat8(*(const v8h*)(arow1 + kb + 16 * hi),
                   *(const v8h*)(arow1 + kb + 16 * hi + 8));
    #pragma unroll
    for (int t = 0; t < 4; ++t) {
      v16h b = *(const v16h*)(Bt + (size_t)(n0 + t * 16 + nn) * K + kb + 16 * hi);
      acc[0][t] = wmma_f16(a0, b, acc[0][t]);
      acc[1][t] = wmma_f16(a1, b, acc[1][t]);
    }
  }
  #pragma unroll
  for (int mi = 0; mi < 2; ++mi)
    #pragma unroll
    for (int t = 0; t < 4; ++t)
      #pragma unroll
      for (int r = 0; r < 8; ++r) {
        int m = m0 + mi * 16 + r + 8 * hi;   // C layout: vgpr r -> row r (+8 upper half)
        int n = n0 + t * 16 + nn;
        if (Ch) Ch[(size_t)m * N + n] = (_Float16)acc[mi][t][r];
        else    Cf[(size_t)m * N + n] = acc[mi][t][r];
      }
}

// ---------------------------------------------------------------- YaRN RoPE
// Q0 [S,2048], K0/V0 [S,512] pre-rope f16 ->
// Qh [H][S][128] (x 1/sqrt(128)), Kh [KVH][S][128], Vt [KVH][128][S]
__global__ void k_rope(const int* __restrict__ pos,
                       const _Float16* __restrict__ Q0, const _Float16* __restrict__ K0,
                       const _Float16* __restrict__ V0,
                       _Float16* __restrict__ Qh, _Float16* __restrict__ Kh,
                       _Float16* __restrict__ Vt) {
  const int s = blockIdx.x;
  const int i = threadIdx.x;            // 0..63 rotary pair index
  float p = (float)pos[s];
  float pf = powf(500000.0f, (float)i * (1.0f / 64.0f));
  float interp = 1.0f / (16.0f * pf);   // scaling = 131072/8192 = 16
  float extrap = 1.0f / pf;
  float sm = ((float)i - 18.0f) * (1.0f / 17.0f);   // low=18, high=35
  sm = fminf(fmaxf(sm, 0.0f), 1.0f);
  float inv = (1.0f - sm) * interp + sm * extrap;
  float fr = p * inv;
  const float MSC = 1.2772588722239781f;            // 0.1*ln(16)+1
  float cs = cosf(fr) * MSC;
  float sn = sinf(fr) * MSC;
  const float qs = 0.08838834764831845f;            // 1/sqrt(128)

  #pragma unroll
  for (int h = 0; h < HEADS; ++h) {
    float x1 = (float)Q0[(size_t)s * HID + h * HD + i];
    float x2 = (float)Q0[(size_t)s * HID + h * HD + i + 64];
    size_t o = ((size_t)h * S_LEN + s) * HD;
    Qh[o + i]      = (_Float16)((x1 * cs - x2 * sn) * qs);
    Qh[o + i + 64] = (_Float16)((x2 * cs + x1 * sn) * qs);
  }
  #pragma unroll
  for (int h = 0; h < KVHEADS; ++h) {
    float x1 = (float)K0[(size_t)s * 512 + h * HD + i];
    float x2 = (float)K0[(size_t)s * 512 + h * HD + i + 64];
    size_t o = ((size_t)h * S_LEN + s) * HD;
    Kh[o + i]      = (_Float16)(x1 * cs - x2 * sn);
    Kh[o + i + 64] = (_Float16)(x2 * cs + x1 * sn);
    Vt[((size_t)h * HD + i) * S_LEN + s]      = V0[(size_t)s * 512 + h * HD + i];
    Vt[((size_t)h * HD + i + 64) * S_LEN + s] = V0[(size_t)s * 512 + h * HD + i + 64];
  }
}

// ---------------------------------------------------------------- flash attention
// block = 8 waves = one head x 128 consecutive query rows (8 tiles of 16).
// The 32-key K/V tiles are staged into LDS once per step (async-to-LDS path)
// and shared by all 8 waves (8x cut in L2->WGP traffic).
__global__ __launch_bounds__(256) void k_attn(const _Float16* __restrict__ Qh,
                                              const _Float16* __restrict__ Kh,
                                              const _Float16* __restrict__ Vt,
                                              _Float16* __restrict__ Ah) {
  __shared__ __align__(128) _Float16 Ktile[32 * 128];   // [key][d]   8 KB
  __shared__ __align__(128) _Float16 Vtile[128 * 32];   // [d][key]   8 KB
  __shared__ __align__(128) _Float16 pbuf[8][512];      // P tiles    8 KB
  const int tid  = threadIdx.x;
  const int lane = tid & 31;
  const int wid  = tid >> 5;
  const int head = blockIdx.x >> 5;                 // 16 heads x 32 blocks
  const int r0b  = (blockIdx.x & 31) * 128;         // block query base
  const int r0   = r0b + wid * 16;                  // wave query tile
  const int kh   = head >> 2;                       // GQA group
  const int nn   = lane & 15;
  const int hi   = lane >> 4;
  _Float16* pb = &pbuf[wid][0];

  // Q A-fragments (d chunks of 32)
  v16h aq[4];
  const _Float16* qrow = Qh + ((size_t)head * S_LEN + (r0 + nn)) * HD;
  #pragma unroll
  for (int c = 0; c < 4; ++c)
    aq[c] = cat8(*(const v8h*)(qrow + 32 * c + 16 * hi),
                 *(const v8h*)(qrow + 32 * c + 16 * hi + 8));

  v8f z = {};
  v8f acc[8];
  #pragma unroll
  for (int t = 0; t < 8; ++t) acc[t] = z;
  float mrow[8], lrow[8];
  #pragma unroll
  for (int r = 0; r < 8; ++r) { mrow[r] = -1e30f; lrow[r] = 0.0f; }

  // staging addresses for this thread (32 B each of K and V per step)
  const int vd = tid >> 1, vp = tid & 1;
  const _Float16* gK0 = Kh + ((size_t)kh * S_LEN) * HD + tid * 16;
  const _Float16* gV0 = Vt + ((size_t)kh * HD + vd) * S_LEN + vp * 16;
  _Float16* lK = Ktile + tid * 16;
  _Float16* lV = Vtile + vd * 32 + vp * 16;

  for (int kb = 0; kb <= r0b + 127; kb += 32) {     // block-uniform causal bound
    // ---- stage K (32x128) and V (128x32) tiles into LDS
    stage32(gK0 + (size_t)kb * HD, lK);
    stage32(gV0 + kb, lV);
    stage_wait();
    __syncthreads();

    if (kb <= r0 + 15) {                            // this wave still in causal range
      v8f c0 = z, c1 = z;
      const _Float16* kp0 = Ktile + nn * HD + 16 * hi;
      const _Float16* kp1 = kp0 + 16 * HD;
      #pragma unroll
      for (int c = 0; c < 4; ++c) c0 = wmma_f16(aq[c], *(const v16h*)(kp0 + 32 * c), c0);
      #pragma unroll
      for (int c = 0; c < 4; ++c) c1 = wmma_f16(aq[c], *(const v16h*)(kp1 + 32 * c), c1);

      if (kb + 31 > r0) {                           // diagonal: causal mask
        #pragma unroll
        for (int r = 0; r < 8; ++r) {
          int qg = r0 + r + 8 * hi;
          if (kb + nn > qg)      c0[r] = -1e30f;
          if (kb + 16 + nn > qg) c1[r] = -1e30f;
        }
      }

      float p0[8], p1[8];
      #pragma unroll
      for (int r = 0; r < 8; ++r) {
        float mx = fmaxf(c0[r], c1[r]);
        for (int m = 1; m < 16; m <<= 1) mx = fmaxf(mx, __shfl_xor(mx, m, 32));
        float mnew  = fmaxf(mrow[r], mx);
        float alpha = __expf(mrow[r] - mnew);
        p0[r] = __expf(c0[r] - mnew);
        p1[r] = __expf(c1[r] - mnew);
        float rs = p0[r] + p1[r];
        for (int m = 1; m < 16; m <<= 1) rs += __shfl_xor(rs, m, 32);
        lrow[r] = lrow[r] * alpha + rs;
        mrow[r] = mnew;
        #pragma unroll
        for (int t = 0; t < 8; ++t) acc[t][r] *= alpha;
      }

      // C-layout -> A-layout via per-wave LDS bounce (DS in-order per wave)
      #pragma unroll
      for (int r = 0; r < 8; ++r) {
        int m = r + 8 * hi;
        pb[m * 32 + nn]      = (_Float16)p0[r];
        pb[m * 32 + 16 + nn] = (_Float16)p1[r];
      }
      __builtin_amdgcn_wave_barrier();
      v16h pa = cat8(*(const v8h*)(pb + nn * 32 + 16 * hi),
                     *(const v8h*)(pb + nn * 32 + 16 * hi + 8));
      __builtin_amdgcn_wave_barrier();

      #pragma unroll
      for (int t = 0; t < 8; ++t) {
        v16h bv = *(const v16h*)(Vtile + (t * 16 + nn) * 32 + 16 * hi);
        acc[t] = wmma_f16(pa, bv, acc[t]);
      }
    }
    __syncthreads();                                // before tiles are overwritten
  }

  #pragma unroll
  for (int r = 0; r < 8; ++r) {
    float recip = 1.0f / (lrow[r] + 1e-10f);
    int m = r0 + r + 8 * hi;
    #pragma unroll
    for (int t = 0; t < 8; ++t)
      Ah[(size_t)m * HID + head * HD + t * 16 + nn] = (_Float16)(acc[t][r] * recip);
  }
}

// ---------------------------------------------------------------- launch
extern "C" void kernel_launch(void* const* d_in, const int* in_sizes, int n_in,
                              void* d_out, int out_size, void* d_ws, size_t ws_size,
                              hipStream_t stream) {
  const float* X   = (const float*)d_in[0];
  const int*   pos = (const int*)d_in[1];
  const float* Wq  = (const float*)d_in[2];
  const float* Wk  = (const float*)d_in[3];
  const float* Wv  = (const float*)d_in[4];
  const float* Wo  = (const float*)d_in[5];
  float* out = (float*)d_out;

  char* w = (char*)d_ws;
  _Float16* Xh  = (_Float16*)(w);                      // 16 MB  [4096,2048]
  _Float16* WqT = (_Float16*)(w + 16777216);           //  8 MB  [2048,2048]
  _Float16* WkT = (_Float16*)(w + 25165824);           //  2 MB  [512,2048]
  _Float16* WvT = (_Float16*)(w + 27262976);           //  2 MB  [512,2048]
  _Float16* WoT = (_Float16*)(w + 29360128);           //  8 MB  [2048,2048]
  _Float16* Q0  = (_Float16*)(w + 37748736);           // 16 MB  [4096,2048] pre-rope
  _Float16* K0  = (_Float16*)(w + 54525952);           //  4 MB  [4096,512]
  _Float16* V0  = (_Float16*)(w + 58720256);           //  4 MB  [4096,512]
  _Float16* Qh  = (_Float16*)(w + 62914560);           // 16 MB  [16][4096][128]
  _Float16* Kh  = (_Float16*)(w + 79691776);           //  4 MB  [4][4096][128]
  _Float16* Vt  = (_Float16*)(w + 83886080);           //  4 MB  [4][128][4096]
  _Float16* Ah  = (_Float16*)(w + 88080384);           // 16 MB  [4096,2048]

  k_f32_to_f16<<<8388608 / 256, 256, 0, stream>>>(X, Xh, 8388608);
  k_transpose_f16<<<(2048 * 2048) / 256, 256, 0, stream>>>(Wq, WqT, 2048, 2048);
  k_transpose_f16<<<(2048 * 512) / 256, 256, 0, stream>>>(Wk, WkT, 2048, 512);
  k_transpose_f16<<<(2048 * 512) / 256, 256, 0, stream>>>(Wv, WvT, 2048, 512);
  k_transpose_f16<<<(2048 * 2048) / 256, 256, 0, stream>>>(Wo, WoT, 2048, 2048);

  k_gemm<<<dim3(32, 32), 128, 0, stream>>>(Xh, WqT, nullptr, Q0, 4096, 2048, 2048);
  k_gemm<<<dim3(32, 8),  128, 0, stream>>>(Xh, WkT, nullptr, K0, 4096, 512, 2048);
  k_gemm<<<dim3(32, 8),  128, 0, stream>>>(Xh, WvT, nullptr, V0, 4096, 512, 2048);

  k_rope<<<4096, 64, 0, stream>>>(pos, Q0, K0, V0, Qh, Kh, Vt);

  k_attn<<<512, 256, 0, stream>>>(Qh, Kh, Vt, Ah);

  k_gemm<<<dim3(32, 32), 128, 0, stream>>>(Ah, WoT, out, nullptr, 4096, 2048, 2048);
}
